// NTXentLoss_35442070126998
// MI455X (gfx1250) — matrix operand order repaired and checked by reading, not compiled, not run on hardware
//
#include <hip/hip_runtime.h>
#include <hip/hip_bf16.h>

typedef __attribute__((ext_vector_type(16))) _Float16 v16h;
typedef __attribute__((ext_vector_type(8)))  _Float16 v8h;
typedef __attribute__((ext_vector_type(4)))  _Float16 v4h;
typedef __attribute__((ext_vector_type(8)))  float    v8f;
typedef __attribute__((ext_vector_type(4)))  float    v4f;
typedef __attribute__((ext_vector_type(4)))  int      v4i;

#define DIMD   128
#define TILE   16
#define SHIFT  40.0f
#define HNW    0.3f
#define MARG   0.3f
#define MCOEF  0.3f
#define NEGBIG (-3.4e38f)

// Structurally branchless selects (and/or/bfi — cannot become control flow)
__device__ __forceinline__ int isel(bool c, int a, int b) {
    int m = -(int)c;
    return (a & m) | (b & ~m);
}
__device__ __forceinline__ float fsel(bool c, float a, float b) {
    return __int_as_float(isel(c, __float_as_int(a), __float_as_int(b)));
}

// ---- fp16 fragment loaders (wave32 WMMA layouts), operands pre-converted ----
// A (16x32, MxK): lane m=lane&15 holds row m; K = 16hs..16hs+15 contiguous
__device__ __forceinline__ v16h load_a_frag(const _Float16* __restrict__ rowp, int kc, int hs) {
    v8h q0 = *(const v8h*)(rowp + kc * 32 + hs * 16);        // 16B aligned
    v8h q1 = *(const v8h*)(rowp + kc * 32 + hs * 16 + 8);
    return __builtin_shufflevector(q0, q1, 0, 1, 2, 3, 4, 5, 6, 7,
                                            8, 9, 10, 11, 12, 13, 14, 15);
}
// B (32x16, KxN): lane m holds column m; K = 16hs..16hs+15 contiguous in S's row
__device__ __forceinline__ v16h load_b_frag(const _Float16* __restrict__ colp, int kc, int hs) {
    return *(const v16h*)(colp + kc * 32 + hs * 16);        // 32B -> 2x b128
}

#define WMMA(C, A, Bm) \
    C = __builtin_amdgcn_wmma_f32_16x16x32_f16(false, (A), false, (Bm), (short)0, (C), false, false)

// ---------------- Pass 0: one-shot fp32 -> fp16 conversion -------------------------
__global__ void __launch_bounds__(256) ntx_cvt(
    const float* __restrict__ in, _Float16* __restrict__ out, int n4) {
    int i = blockIdx.x * blockDim.x + threadIdx.x;
    if (i < n4) {
        v4f q = ((const v4f*)in)[i];
        v4h h;
        h[0] = (_Float16)q[0]; h[1] = (_Float16)q[1];
        h[2] = (_Float16)q[2]; h[3] = (_Float16)q[3];
        ((v4h*)out)[i] = h;
    }
}

// ---------------- Pass 1: row stats (top-2, diag, boosted row denominator) ---------
__global__ void __launch_bounds__(256) ntx_rowpass(
    const _Float16* __restrict__ I16, const _Float16* __restrict__ S16,
    int* __restrict__ h1o, int* __restrict__ h2o,
    float* __restrict__ v1o, float* __restrict__ v2o,
    float* __restrict__ diago, float* __restrict__ rdeno, int B) {
    const int lane = threadIdx.x & 31;
    const int wave = threadIdx.x >> 5;
    const int m    = lane & 15;
    const int hs   = lane >> 4;
    const int row_base = (blockIdx.x * 8 + wave) * TILE;
    if (row_base >= B) return;  // wave-uniform

    const _Float16* rowp = I16 + (size_t)(row_base + m) * DIMD;
    v16h a0 = load_a_frag(rowp, 0, hs);
    v16h a1 = load_a_frag(rowp, 1, hs);
    v16h a2 = load_a_frag(rowp, 2, hs);
    v16h a3 = load_a_frag(rowp, 3, hs);

    float t1v[8], t2v[8], sumv[8], dgv[8];
    int   t1i[8], t2i[8];
#pragma unroll
    for (int v = 0; v < 8; ++v) {
        t1v[v] = NEGBIG; t2v[v] = NEGBIG;
        t1i[v] = 0; t2i[v] = 0; sumv[v] = 0.f; dgv[v] = 0.f;
    }

    // per-tile epilogue: online sum-of-exp, diag capture, branchless top-2
    auto tile_update = [&](int cc, const v8f& c) {
#pragma unroll
        for (int v = 0; v < 8; ++v) {
            const int   r  = row_base + hs * 8 + v;
            const float x  = c[v];
            const bool  dg = (cc == r);
            sumv[v] += __expf(x - SHIFT);
            dgv[v]   = fsel(dg, x, dgv[v]);
            const float xm = fsel(dg, NEGBIG, x);   // diagonal masked out
            const bool  g1 = xm > t1v[v];
            const bool  g2 = xm > t2v[v];
            t2i[v] = isel(g1, t1i[v], isel(g2, cc, t2i[v]));
            t2v[v] = fmaxf(t2v[v], fminf(t1v[v], xm));
            t1i[v] = isel(g1, cc, t1i[v]);
            t1v[v] = fmaxf(t1v[v], xm);
        }
    };

    const int nct = B / TILE;   // 512 (even)
    // ping-pong fragment sets: no rotation copies, loads run 1 tile ahead
    const _Float16* colp = S16 + (size_t)m * DIMD;
    v16h pA0 = load_b_frag(colp, 0, hs);
    v16h pA1 = load_b_frag(colp, 1, hs);
    v16h pA2 = load_b_frag(colp, 2, hs);
    v16h pA3 = load_b_frag(colp, 3, hs);

    for (int ct = 0; ct < nct; ct += 2) {
        // load tile ct+1 (always valid) into set B while computing tile ct
        const _Float16* c1 = S16 + (size_t)((ct + 1) * TILE + m) * DIMD;
        v16h pB0 = load_b_frag(c1, 0, hs);
        v16h pB1 = load_b_frag(c1, 1, hs);
        v16h pB2 = load_b_frag(c1, 2, hs);
        v16h pB3 = load_b_frag(c1, 3, hs);
        {
            v8f c = {};
            WMMA(c, a0, pA0); WMMA(c, a1, pA1); WMMA(c, a2, pA2); WMMA(c, a3, pA3);
            tile_update(ct * TILE + m, c);
        }
        // load tile ct+2 (clamped, branchless) into set A while computing tile ct+1
        const int ct2 = isel(ct + 2 < nct, ct + 2, 0);
        const _Float16* c2 = S16 + (size_t)(ct2 * TILE + m) * DIMD;
        pA0 = load_b_frag(c2, 0, hs);
        pA1 = load_b_frag(c2, 1, hs);
        pA2 = load_b_frag(c2, 2, hs);
        pA3 = load_b_frag(c2, 3, hs);
        {
            v8f c = {};
            WMMA(c, a0, pB0); WMMA(c, a1, pB1); WMMA(c, a2, pB2); WMMA(c, a3, pB3);
            tile_update((ct + 1) * TILE + m, c);
        }
    }

    // butterfly all-reduce across the 16 lanes sharing hs (xor 1,2,4,8 stay in group)
#pragma unroll
    for (int v = 0; v < 8; ++v) {
#pragma unroll
        for (int off = 1; off < 16; off <<= 1) {
            const float b1s = __shfl_xor(t1v[v], off, 32);
            const int   bi1 = __shfl_xor(t1i[v], off, 32);
            const float b2s = __shfl_xor(t2v[v], off, 32);
            const int   bi2 = __shfl_xor(t2i[v], off, 32);
            sumv[v] += __shfl_xor(sumv[v], off, 32);
            dgv[v]  += __shfl_xor(dgv[v], off, 32);  // exactly one lane holds nonzero
            const bool  bw  = (b1s > t1v[v]);
            const float lo  = fminf(t1v[v], b1s);         // loser of the firsts
            const int   loi = isel(bw, t1i[v], bi1);
            const float wsv = fsel(bw, b2s, t2v[v]);      // second of the winner pair
            const int   wsi = isel(bw, bi2, t2i[v]);
            const bool  s2  = (lo > wsv);
            t1i[v] = isel(bw, bi1, t1i[v]);
            t1v[v] = fmaxf(t1v[v], b1s);
            t2v[v] = fmaxf(lo, wsv);
            t2i[v] = isel(s2, loi, wsi);
        }
        if (m == 0) {
            const int r = row_base + hs * 8 + v;
            const float v1 = t1v[v], v2 = t2v[v];
            const float den = sumv[v] - __expf(v1 - SHIFT) - __expf(v2 - SHIFT)
                            + __expf(v1 * (1.f + HNW) - SHIFT)
                            + __expf(v2 * (1.f + HNW * 0.5f) - SHIFT);
            h1o[r] = t1i[v]; h2o[r] = t2i[v];
            v1o[r] = v1;     v2o[r] = v2;
            diago[r] = dgv[v]; rdeno[r] = den;
        }
    }
}

// ---------------- Pass 2: boosted column denominators ------------------------------
__global__ void __launch_bounds__(256) ntx_colpass(
    const _Float16* __restrict__ I16, const _Float16* __restrict__ S16,
    const int* __restrict__ h1, const int* __restrict__ h2,
    float* __restrict__ colsum, int B) {
    const int lane = threadIdx.x & 31;
    const int wave = threadIdx.x >> 5;
    const int m    = lane & 15;
    const int hs   = lane >> 4;
    const int col_base = (blockIdx.x * 8 + wave) * TILE;
    if (col_base >= B) return;

    const _Float16* colp = S16 + (size_t)(col_base + m) * DIMD;
    v16h b0 = load_b_frag(colp, 0, hs);
    v16h b1 = load_b_frag(colp, 1, hs);
    v16h b2 = load_b_frag(colp, 2, hs);
    v16h b3 = load_b_frag(colp, 3, hs);

    const int cc = col_base + m;
    float acc = 0.f;

    auto col_update = [&](const v8f& c, v4i ha, v4i hb, v4i ga, v4i gb) {
#pragma unroll
        for (int v = 0; v < 8; ++v) {
            const float x   = c[v];
            const int   hr1 = (v < 4) ? ha[v & 3] : hb[v & 3];
            const int   hr2 = (v < 4) ? ga[v & 3] : gb[v & 3];
            const float f = fsel(cc == hr1, 1.f + HNW,
                            fsel(cc == hr2, 1.f + HNW * 0.5f, 1.f));
            acc += __expf(x * f - SHIFT);
        }
    };

    const int nrt = B / TILE;   // 512 (even)
    // ping-pong A-fragment + hard-neg-index sets
    const _Float16* rp = I16 + (size_t)m * DIMD;
    v16h aA0 = load_a_frag(rp, 0, hs);
    v16h aA1 = load_a_frag(rp, 1, hs);
    v16h aA2 = load_a_frag(rp, 2, hs);
    v16h aA3 = load_a_frag(rp, 3, hs);
    v4i h1aA = ((const v4i*)(h1 + hs * 8))[0], h1bA = ((const v4i*)(h1 + hs * 8))[1];
    v4i h2aA = ((const v4i*)(h2 + hs * 8))[0], h2bA = ((const v4i*)(h2 + hs * 8))[1];

    for (int rt = 0; rt < nrt; rt += 2) {
        // load row tile rt+1 into set B
        const _Float16* r1 = I16 + (size_t)((rt + 1) * TILE + m) * DIMD;
        v16h aB0 = load_a_frag(r1, 0, hs);
        v16h aB1 = load_a_frag(r1, 1, hs);
        v16h aB2 = load_a_frag(r1, 2, hs);
        v16h aB3 = load_a_frag(r1, 3, hs);
        const int ro1 = (rt + 1) * TILE + hs * 8;
        v4i h1aB = ((const v4i*)(h1 + ro1))[0], h1bB = ((const v4i*)(h1 + ro1))[1];
        v4i h2aB = ((const v4i*)(h2 + ro1))[0], h2bB = ((const v4i*)(h2 + ro1))[1];
        {
            v8f c = {};
            WMMA(c, aA0, b0); WMMA(c, aA1, b1); WMMA(c, aA2, b2); WMMA(c, aA3, b3);
            col_update(c, h1aA, h1bA, h2aA, h2bA);
        }
        // load row tile rt+2 (clamped, branchless) into set A
        const int rt2 = isel(rt + 2 < nrt, rt + 2, 0);
        const _Float16* r2 = I16 + (size_t)(rt2 * TILE + m) * DIMD;
        aA0 = load_a_frag(r2, 0, hs);
        aA1 = load_a_frag(r2, 1, hs);
        aA2 = load_a_frag(r2, 2, hs);
        aA3 = load_a_frag(r2, 3, hs);
        const int ro2 = rt2 * TILE + hs * 8;
        h1aA = ((const v4i*)(h1 + ro2))[0]; h1bA = ((const v4i*)(h1 + ro2))[1];
        h2aA = ((const v4i*)(h2 + ro2))[0]; h2bA = ((const v4i*)(h2 + ro2))[1];
        {
            v8f c = {};
            WMMA(c, aB0, b0); WMMA(c, aB1, b1); WMMA(c, aB2, b2); WMMA(c, aB3, b3);
            col_update(c, h1aB, h1bB, h2aB, h2bB);
        }
    }
    acc += __shfl_xor(acc, 16, 32);  // fold the two row halves of this column
    if (hs == 0) colsum[cc] = acc;
}

// ---------------- Pass 3: scalar reduction -----------------------------------------
__global__ void __launch_bounds__(256) ntx_final(
    const float* __restrict__ v1, const float* __restrict__ diag,
    const float* __restrict__ rden, const float* __restrict__ csum,
    float* __restrict__ out, int B) {
    __shared__ float s0[256], s1[256], s2[256];
    const int tid = threadIdx.x;
    float ai = 0.f, as = 0.f, am = 0.f;
    for (int r = tid; r < B; r += 256) {
        float d = diag[r];
        ai += (d - SHIFT) - __logf(rden[r]);  // log_softmax(boosted)[r,r]
        as += (d - SHIFT) - __logf(csum[r]);  // log_softmax(boosted.T)[r,r]
        am += fmaxf(v1[r] - d + MARG, 0.f);
    }
    s0[tid] = ai; s1[tid] = as; s2[tid] = am;
    __syncthreads();
    for (int s = 128; s > 0; s >>= 1) {
        if (tid < s) { s0[tid] += s0[tid + s]; s1[tid] += s1[tid + s]; s2[tid] += s2[tid + s]; }
        __syncthreads();
    }
    if (tid == 0) {
        float invB = 1.f / (float)B;
        float li2s = -s0[0] * invB;
        float ls2i = -s1[0] * invB;
        float lm   =  s2[0] * invB;
        out[0] = 0.5f * (li2s + ls2i) + MCOEF * lm;
    }
}

extern "C" void kernel_launch(void* const* d_in, const int* in_sizes, int n_in,
                              void* d_out, int out_size, void* d_ws, size_t ws_size,
                              hipStream_t stream) {
    const float* I = (const float*)d_in[0];
    const float* S = (const float*)d_in[1];
    const int B = in_sizes[0] / DIMD;       // 8192
    const int nElem = B * DIMD;             // 1,048,576 per matrix

    // Workspace layout: fp16 copies first (4 MB), then per-row/col stats (224 KB)
    _Float16* I16 = (_Float16*)d_ws;
    _Float16* S16 = I16 + nElem;
    float* ws   = (float*)(S16 + nElem);
    int*   h1   = (int*)(ws);
    int*   h2   = (int*)(ws + B);
    float* v1   = ws + 2 * B;
    float* v2   = ws + 3 * B;
    float* diag = ws + 4 * B;
    float* rden = ws + 5 * B;
    float* csum = ws + 6 * B;

    const int n4 = nElem / 4;
    const int cvtBlocks = (n4 + 255) / 256;
    ntx_cvt<<<cvtBlocks, 256, 0, stream>>>(I, I16, n4);
    ntx_cvt<<<cvtBlocks, 256, 0, stream>>>(S, S16, n4);

    const int tilesPerBlock = 8;                    // 8 waves / 256-thread block
    const int blocks = (B / TILE) / tilesPerBlock;  // 64

    ntx_rowpass<<<blocks, 256, 0, stream>>>(I16, S16, h1, h2, v1, v2, diag, rden, B);
    ntx_colpass<<<blocks, 256, 0, stream>>>(I16, S16, h1, h2, csum, B);
    ntx_final<<<1, 256, 0, stream>>>(v1, diag, rden, csum, (float*)d_out, B);
}
